// TranscriptomePredictor_75479755260448
// MI455X (gfx1250) — matrix operand order repaired and checked by reading, compiled, not run on hardware
//
#include <hip/hip_runtime.h>

// ---------- types ----------
typedef __bf16 bf16_t;
typedef __attribute__((ext_vector_type(8)))  __bf16 bf16x8;
typedef __attribute__((ext_vector_type(16))) __bf16 bf16x16;
typedef __attribute__((ext_vector_type(8)))  float  f32x8;
typedef __attribute__((ext_vector_type(4)))  unsigned int u32x4;
typedef __attribute__((ext_vector_type(8)))  int i32x8;
typedef __attribute__((ext_vector_type(4)))  int i32x4;

// ---------- model constants ----------
#define LSEQ   8193            // 1 cond + 8192 genes
#define LHAT   8320            // padded to 65 chunks of 128
#define NCHUNK 65
#define QCH    128             // chunk length
#define DMODEL 256
#define GFDP   192             // GFD=178 padded to 192
#define HH     8
#define PP     64
#define DI     512             // H*P
#define NSD    128
#define DINP   1312            // 2*512+2*128+8 = 1288, padded to mult of 32
#define DINR   1288
#define CONVD  768
#define NTOK   8224            // 8192 genes + 8 cond rows + pad to mult of 32

// ---------------------------------------------------------------
// WMMA fragment loads (bf16 16x16x32):
// A-frag: lane<16 -> k {0..15};       lane>=16 -> k {16..31}
// B-frag: lane<16 -> k {0..15};       lane>=16 -> k {16..31}
// ---------------------------------------------------------------
__device__ inline bf16x16 ld_afrag(const bf16_t* __restrict__ p, int kk)
{
    bf16x8 lo = *(const bf16x8*)(p + kk);
    bf16x8 hi = *(const bf16x8*)(p + kk + 8);
    return __builtin_shufflevector(lo, hi, 0,1,2,3,4,5,6,7,8,9,10,11,12,13,14,15);
}
__device__ inline bf16x16 ld_bfrag(const bf16_t* __restrict__ p, int kk)
{
    bf16x8 lo = *(const bf16x8*)(p + kk);
    bf16x8 hi = *(const bf16x8*)(p + kk + 8);
    return __builtin_shufflevector(lo, hi, 0,1,2,3,4,5,6,7,8,9,10,11,12,13,14,15);
}
__device__ inline f32x8 wmma_bf16(bf16x16 a, bf16x16 b, f32x8 c)
{
    return __builtin_amdgcn_wmma_f32_16x16x32_bf16(false, a, false, b,
                                                   (short)0, c, false, false);
}

// 2x2 register macro-tile: C(32x32) = A(32xK) * BT(32xK)^T.
// A row-major [m][k]; B operand row-major [n][k] (global BT or LDS panel).
template<bool PFA, bool PFB>
__device__ inline void mac2x2(const bf16_t* __restrict__ A, long lda,
                              const bf16_t* __restrict__ BT, long ldb,
                              int K, f32x8 (&acc)[2][2])
{
    int lane = threadIdx.x & 31;
    int rsel = lane & 15;
    const bf16_t* a0 = A  + (long)rsel * lda + ((lane >> 4) << 4);
    const bf16_t* a1 = a0 + 16 * lda;
    const bf16_t* b0 = BT + (long)rsel * ldb + ((lane >> 4) << 4);
    const bf16_t* b1 = b0 + 16 * ldb;
    for (int kk = 0; kk < K; kk += 32) {
        if (kk + 128 < K) {                       // GL2 warm for long-K GEMMs
            if (PFA) __builtin_prefetch(a0 + kk + 128, 0, 0);
            if (PFB) __builtin_prefetch(b0 + kk + 128, 0, 0);
        }
        bf16x16 af0 = ld_afrag(a0, kk);
        bf16x16 af1 = ld_afrag(a1, kk);
        bf16x16 bf0 = ld_bfrag(b0, kk);
        bf16x16 bf1 = ld_bfrag(b1, kk);
        acc[0][0] = wmma_bf16(af0, bf0, acc[0][0]);
        acc[0][1] = wmma_bf16(af0, bf1, acc[0][1]);
        acc[1][0] = wmma_bf16(af1, bf0, acc[1][0]);
        acc[1][1] = wmma_bf16(af1, bf1, acc[1][1]);
    }
}
__device__ inline void zero_acc(f32x8 (&acc)[2][2])
{
    #pragma unroll
    for (int i = 0; i < 2; ++i)
        #pragma unroll
        for (int j = 0; j < 2; ++j)
            acc[i][j] = (f32x8){0.f,0.f,0.f,0.f,0.f,0.f,0.f,0.f};
}

__device__ inline int tok_index(int dir, int b, int p)
{
    if (dir == 0) return (p == 0) ? (8192 + b) : (p - 1);
    return (p < 8192) ? (8191 - p) : (8192 + b);
}

// ---------------------------------------------------------------
// Feature assembly: rows 0..8191 genes, 8192..8199 cond(b), rest 0.
// ---------------------------------------------------------------
__global__ void k_build_feats(const float* __restrict__ gene_emb,
                              const float* __restrict__ chr_emb,
                              const float* __restrict__ locus_W,
                              const float* __restrict__ locus_b,
                              const float* __restrict__ pert_emb,
                              const float* __restrict__ cond_W,
                              const float* __restrict__ cond_b,
                              const int*   __restrict__ chr_idx,
                              const float* __restrict__ locus_fourier,
                              const float* __restrict__ pathway,
                              const int*   __restrict__ pert_idx,
                              bf16_t* __restrict__ featsB)
{
    int row = blockIdx.x;
    for (int j = threadIdx.x; j < GFDP; j += 64) {
        float v = 0.f;
        if (row < 8192) {
            int g = row;
            if (j < 64)        v = gene_emb[g * 64 + j];
            else if (j < 114)  v = pathway[g * 50 + (j - 64)];
            else if (j < 146)  v = chr_emb[chr_idx[g] * 32 + (j - 114)];
            else if (j < 178) {
                int jj = j - 146;
                float s = locus_b[jj];
                for (int k = 0; k < 32; ++k)
                    s += locus_fourier[g * 32 + k] * locus_W[k * 32 + jj];
                v = 0.5f * s * (1.f + erff(s * 0.70710678118f));  // exact gelu
            }
        } else if (row < 8200) {
            int b = row - 8192;
            if (j < 178) {
                float s = cond_b[j];
                int pi = pert_idx[b];
                for (int k = 0; k < 128; ++k)
                    s += pert_emb[pi * 128 + k] * cond_W[k * 178 + j];
                v = s;
            }
        }
        featsB[(long)row * GFDP + j] = (bf16_t)v;
    }
}

// Transpose+pad fp32 (K x N) -> bf16 (Npad x Kpad), zero outside.
__global__ void k_transpose_pad(const float* __restrict__ src, bf16_t* __restrict__ dst,
                                int K, int N, int Kpad, int Npad)
{
    long idx = (long)blockIdx.x * blockDim.x + threadIdx.x;
    long total = (long)Kpad * Npad;
    if (idx >= total) return;
    int k = (int)(idx % Kpad);
    int n = (int)(idx / Kpad);
    float v = (k < K && n < N) ? src[(long)k * N + n] : 0.f;
    dst[idx] = (bf16_t)v;
}

// ---------------------------------------------------------------
// Generic WMMA GEMM (2x2 macro-tiles, M,N mult of 32), 2D grid:
// blockIdx.y = N macro-tile (shared by the whole block), blockIdx.x*8+wave = M
// macro-tile.  useLds: one TDM tensor_load_to_lds stages the 32xK B-panel
// (+16B/row LDS pad via D# pad fields) once per block; fragments then read
// from LDS.  Requires K*2 bytes to be a power of two when useLds=1.
// mapMode 0: out row = m (valid m<Lvalid). mapMode 1: out row = Lvalid-1-m.
// ---------------------------------------------------------------
__global__ void k_gemm(const bf16_t* __restrict__ A, long lda,
                       const bf16_t* __restrict__ BT, long ldb,
                       const float* __restrict__ bias,
                       float* __restrict__ outF, long ldc,
                       bf16_t* __restrict__ outB, long ldcb,
                       int M, int N, int K, int mapMode, int Lvalid, int addOut,
                       int useLds)
{
    extern __shared__ bf16_t Bs[];
    int numMt = M >> 5;
    int nt = blockIdx.y;
    int ldbs = K + 8;                       // +16B per row (TDM pad)
    if (useLds) {
        if ((threadIdx.x >> 5) == 0) {
            // ---- Tensor DMA descriptor (D#): 2D tile 32 rows x K bf16 ----
            unsigned ldsBase = (unsigned)__builtin_amdgcn_groupstaticsize();
            unsigned long long ga =
                (unsigned long long)(const void*)(BT + (long)nt * 32 * ldb);
            int icode = __builtin_ctz((unsigned)(K >> 1)) - 1;  // 8B units per row
            u32x4 g0;
            g0[0] = 1u;                                          // count=1 (user load)
            g0[1] = ldsBase;                                     // lds_addr
            g0[2] = (unsigned)(ga & 0xffffffffu);                // global_addr lo
            g0[3] = (unsigned)((ga >> 32) & 0x01ffffffu) | 0x80000000u; // hi | type=2
            i32x8 g1;
            g1[0] = (1 << 16) | (1 << 20) | (icode << 22) | (3 << 25);
                     // data_size=2B | pad_enable | pad_interval | pad_amount=4dw
            g1[1] = (int)(((unsigned)ldb & 0xffffu) << 16);      // tensor_dim0 lo16
            g1[2] = (int)((((unsigned)ldb >> 16) & 0xffffu) | (32u << 16));
                     // tensor_dim0 hi16 | tensor_dim1 lo16 = 32
            g1[3] = (int)(((unsigned)K & 0xffffu) << 16);        // tile_dim0 = K
            g1[4] = 32;                                          // tile_dim1=32, tile_dim2=0
            g1[5] = (int)ldb;                                    // tensor_dim0_stride lo
            g1[6] = 0;
            g1[7] = 0;
            i32x4 gz  = {0, 0, 0, 0};
            i32x8 gz8 = {0, 0, 0, 0, 0, 0, 0, 0};
            __builtin_amdgcn_tensor_load_to_lds(g0, g1, gz, gz, gz8, 0);
            __builtin_amdgcn_s_wait_tensorcnt(0);
        }
        __syncthreads();
    }
    int wave = threadIdx.x >> 5;
    int mt = blockIdx.x * (blockDim.x >> 5) + wave;
    if (mt >= numMt) return;
    f32x8 acc[2][2];
    zero_acc(acc);
    if (useLds)
        mac2x2<true, false>(A + (long)mt * 32 * lda, lda, Bs, ldbs, K, acc);
    else
        mac2x2<true, true>(A + (long)mt * 32 * lda, lda,
                           BT + (long)nt * 32 * ldb, ldb, K, acc);
    int lane = threadIdx.x & 31;
    int coln = lane & 15, mhi = (lane >> 4) << 3;
    #pragma unroll
    for (int ti = 0; ti < 2; ++ti) {
        #pragma unroll
        for (int tj = 0; tj < 2; ++tj) {
            int n = nt * 32 + tj * 16 + coln;
            float bs = bias ? bias[n] : 0.f;
            #pragma unroll
            for (int r = 0; r < 8; ++r) {
                int m = mt * 32 + ti * 16 + mhi + r;
                if (m >= Lvalid) continue;
                int t = mapMode ? (Lvalid - 1 - m) : m;
                float v = acc[ti][tj][r] + bs;
                if (outF) {
                    long o = (long)t * ldc + n;
                    if (addOut) v += outF[o];
                    outF[o] = v;
                }
                if (outB) outB[(long)t * ldcb + n] = (bf16_t)v;
            }
        }
    }
}

// Batched plain-f32-store WMMA GEMM; batch offsets from grid.y (head) / grid.z (chunk)
__global__ void k_gemm_batched(const bf16_t* __restrict__ A, long lda, long sAy, long sAz,
                               const bf16_t* __restrict__ BT, long ldb, long sBy, long sBz,
                               float* __restrict__ C, long ldc, long sCy, long sCz,
                               int M, int N, int K)
{
    A  += (long)blockIdx.y * sAy + (long)blockIdx.z * sAz;
    BT += (long)blockIdx.y * sBy + (long)blockIdx.z * sBz;
    C  += (long)blockIdx.y * sCy + (long)blockIdx.z * sCz;
    int wave = threadIdx.x >> 5;
    int tile = blockIdx.x * (blockDim.x >> 5) + wave;
    int numMt = M >> 5, numNt = N >> 5;
    if (tile >= numMt * numNt) return;
    int mt = tile % numMt, nt = tile / numMt;
    f32x8 acc[2][2];
    zero_acc(acc);
    mac2x2<true, true>(A + (long)mt * 32 * lda, lda, BT + (long)nt * 32 * ldb, ldb, K, acc);
    int lane = threadIdx.x & 31;
    int coln = lane & 15, mhi = (lane >> 4) << 3;
    #pragma unroll
    for (int ti = 0; ti < 2; ++ti)
        #pragma unroll
        for (int tj = 0; tj < 2; ++tj)
            #pragma unroll
            for (int r = 0; r < 8; ++r)
                C[(long)(mt * 32 + ti * 16 + mhi + r) * ldc
                  + nt * 32 + tj * 16 + coln] = acc[ti][tj][r];
}

// ---------------------------------------------------------------
// Depthwise conv(4) + SiLU + split into x / B / C, plus dt & a = dt*(-e^A).
// ---------------------------------------------------------------
__global__ void k_conv_split(const float* __restrict__ Z,
                             const float* __restrict__ conv_w,
                             const float* __restrict__ conv_b,
                             const float* __restrict__ dt_bias,
                             const float* __restrict__ A_log,
                             float* __restrict__ x,
                             bf16_t* __restrict__ Bmb,
                             bf16_t* __restrict__ Cmb,
                             bf16_t* __restrict__ BmT,
                             float* __restrict__ dtb,
                             float* __restrict__ ab,
                             int dir, int b)
{
    int p = blockIdx.x;
    bool valid = p < LSEQ;
    for (int c = threadIdx.x; c < CONVD; c += 256) {
        float v = 0.f;
        if (valid) {
            float s = conv_b[c];
            #pragma unroll
            for (int k = 0; k < 4; ++k) {
                int q = p - 3 + k;
                if (q >= 0) {
                    long tok = tok_index(dir, b, q);
                    s += Z[tok * DINP + 512 + c] * conv_w[c * 4 + k];
                }
            }
            v = s / (1.f + __expf(-s));   // silu
        }
        if (c < 512)       x[(long)p * 512 + c] = v;
        else if (c < 640) {
            Bmb[(long)p * NSD + (c - 512)] = (bf16_t)v;
            BmT[(long)(c - 512) * LHAT + p] = (bf16_t)v;
        } else             Cmb[(long)p * NSD + (c - 640)] = (bf16_t)v;
    }
    if (threadIdx.x < HH) {
        int h = threadIdx.x;
        float dt = 0.f, a = 0.f;
        if (valid) {
            long tok = tok_index(dir, b, p);
            float r = Z[tok * DINP + 1280 + h] + dt_bias[h];
            dt = (r > 20.f) ? r : log1pf(__expf(r));      // softplus
            a  = dt * (-__expf(A_log[h]));
        }
        dtb[p * HH + h] = dt;
        ab[p * HH + h]  = a;
    }
}

// Inclusive cumsum of a within each chunk (per head). LDS Hillis-Steele.
__global__ void k_cum_a(const float* __restrict__ ab,
                        float* __restrict__ cumA, float* __restrict__ csum)
{
    int c = blockIdx.x >> 3, h = blockIdx.x & 7;
    int t = threadIdx.x;
    __shared__ float s[QCH];
    int p = c * QCH + t;
    s[t] = ab[p * HH + h];
    __syncthreads();
    for (int off = 1; off < QCH; off <<= 1) {
        float v = s[t] + ((t >= off) ? s[t - off] : 0.f);
        __syncthreads();
        s[t] = v;
        __syncthreads();
    }
    cumA[p * HH + h] = s[t];
    if (t == QCH - 1) csum[c * HH + h] = s[t];
}

// Decay-weighted transposed x layouts: xdtT[idx][p] = x*dt, xwT[idx][p] = x*dt*exp(sum-cum)
__global__ void k_weight_x(const float* __restrict__ x, const float* __restrict__ dtb,
                           const float* __restrict__ cumA, const float* __restrict__ csum,
                           bf16_t* __restrict__ xdtT, bf16_t* __restrict__ xwT)
{
    int p = blockIdx.x, c = p >> 7;
    for (int idx = threadIdx.x; idx < DI; idx += 256) {
        int h = idx >> 6;
        float dt = dtb[p * HH + h];
        float wS = __expf(csum[c * HH + h] - cumA[p * HH + h]) * dt;
        float xv = x[(long)p * DI + idx];
        xdtT[(long)idx * LHAT + p] = (bf16_t)(xv * dt);
        xwT [(long)idx * LHAT + p] = (bf16_t)(xv * wS);
    }
}

// G-matrix: G~[c,h][i][j] = (C_i . B_j) * exp(cum_i - cum_j) * dt_j for j<=i, else 0
__global__ void k_gmat(const bf16_t* __restrict__ Cm, const bf16_t* __restrict__ Bm,
                       const float* __restrict__ cumA, const float* __restrict__ dtb,
                       bf16_t* __restrict__ Gt)
{
    int h = blockIdx.y, c = blockIdx.z;
    int wave = threadIdx.x >> 5;
    int tile = blockIdx.x * 8 + wave;         // 16 macro-tiles (4x4)
    if (tile >= 16) return;
    int mt = tile & 3, nt = tile >> 2;
    const bf16_t* A = Cm + ((long)c * QCH + mt * 32) * NSD;
    const bf16_t* B = Bm + ((long)c * QCH + nt * 32) * NSD;
    f32x8 acc[2][2];
    zero_acc(acc);
    mac2x2<false, false>(A, NSD, B, NSD, NSD, acc);
    int lane = threadIdx.x & 31;
    int coln = lane & 15, mhi = (lane >> 4) << 3;
    #pragma unroll
    for (int tj = 0; tj < 2; ++tj) {
        int jl = nt * 32 + tj * 16 + coln;
        int jg = c * QCH + jl;
        float cj = cumA[jg * HH + h];
        float dj = dtb[jg * HH + h];
        #pragma unroll
        for (int ti = 0; ti < 2; ++ti) {
            #pragma unroll
            for (int r = 0; r < 8; ++r) {
                int il = mt * 32 + ti * 16 + mhi + r;
                int ig = c * QCH + il;
                float w = (jl <= il) ? __expf(cumA[ig * HH + h] - cj) * dj : 0.f;
                Gt[(((long)c * HH + h) * QCH + il) * QCH + jl] = (bf16_t)(acc[ti][tj][r] * w);
            }
        }
    }
}

// Inter-chunk state recurrence: Hprev[c] = state before chunk c (bf16 out for GEMM)
__global__ void k_recur(const float* __restrict__ Sloc, const float* __restrict__ csum,
                        bf16_t* __restrict__ HprevB)
{
    int h = blockIdx.x, t = threadIdx.x;
    float st[32];
    #pragma unroll
    for (int e = 0; e < 32; ++e) st[e] = 0.f;
    for (int c = 0; c < NCHUNK; ++c) {
        long base = ((long)c * HH + h) * (PP * NSD);
        float dec = __expf(csum[c * HH + h]);
        #pragma unroll
        for (int e = 0; e < 32; ++e) {
            int el = e * 256 + t;
            HprevB[base + el] = (bf16_t)st[e];
            st[e] = dec * st[e] + Sloc[base + el];
        }
    }
}

// Y_inter: y[i][h*64+p] += exp(cum_i) * sum_n C[i][n]*Hprev[p][n]
__global__ void k_inter(const bf16_t* __restrict__ Cm, const bf16_t* __restrict__ HprevB,
                        const float* __restrict__ cumA, float* __restrict__ y)
{
    int h = blockIdx.y, c = blockIdx.z;
    int tile = threadIdx.x >> 5;              // 8 macro-tiles (4 mt x 2 nt)
    int mt = tile & 3, nt = tile >> 2;
    const bf16_t* A = Cm + ((long)c * QCH + mt * 32) * NSD;
    const bf16_t* B = HprevB + (((long)c * HH + h) * PP + nt * 32) * NSD;
    f32x8 acc[2][2];
    zero_acc(acc);
    mac2x2<false, false>(A, NSD, B, NSD, NSD, acc);
    int lane = threadIdx.x & 31;
    int coln = lane & 15, mhi = (lane >> 4) << 3;
    #pragma unroll
    for (int ti = 0; ti < 2; ++ti) {
        #pragma unroll
        for (int tj = 0; tj < 2; ++tj) {
            int pdim = nt * 32 + tj * 16 + coln;
            #pragma unroll
            for (int r = 0; r < 8; ++r) {
                int ig = c * QCH + mt * 32 + ti * 16 + mhi + r;
                float sc = __expf(cumA[ig * HH + h]);
                y[(long)ig * DI + h * PP + pdim] += acc[ti][tj][r] * sc;
            }
        }
    }
}

// Gate + RMSNorm: g = (y + D*x) * silu(z); gn = g * rsqrt(mean g^2 + eps) * norm_w
__global__ void k_gate(const float* __restrict__ y, const float* __restrict__ x,
                       const float* __restrict__ Z, const float* __restrict__ Dp,
                       const float* __restrict__ norm_w, bf16_t* __restrict__ gn,
                       int dir, int b)
{
    int p = blockIdx.x, t = threadIdx.x;
    __shared__ float red[256];
    float g0 = 0.f, g1 = 0.f;
    int i0 = t, i1 = t + 256;
    if (p < LSEQ) {
        long tok = tok_index(dir, b, p);
        const float* zr = Z + tok * DINP;
        float yv0 = y[(long)p * DI + i0] + Dp[i0 >> 6] * x[(long)p * DI + i0];
        float z0 = zr[i0];
        g0 = yv0 * (z0 / (1.f + __expf(-z0)));
        float yv1 = y[(long)p * DI + i1] + Dp[i1 >> 6] * x[(long)p * DI + i1];
        float z1 = zr[i1];
        g1 = yv1 * (z1 / (1.f + __expf(-z1)));
    }
    red[t] = g0 * g0 + g1 * g1;
    __syncthreads();
    for (int s = 128; s > 0; s >>= 1) {
        if (t < s) red[t] += red[t + s];
        __syncthreads();
    }
    float rr = rsqrtf(red[0] * (1.f / (float)DI) + 1e-5f);
    gn[(long)p * DI + i0] = (bf16_t)(g0 * rr * norm_w[i0]);
    gn[(long)p * DI + i1] = (bf16_t)(g1 * rr * norm_w[i1]);
}

// Final head: out[b][g] = Hc[b][1+g] . head_W + head_b   (one warp per row)
__global__ void k_head(const float* __restrict__ Hc, const float* __restrict__ head_W,
                       const float* __restrict__ head_b, float* __restrict__ out)
{
    int warp = threadIdx.x >> 5, lane = threadIdx.x & 31;
    long row = (long)blockIdx.x * 8 + warp;       // 0 .. 65535
    int b = (int)(row >> 13), g = (int)(row & 8191);
    const float* hr = Hc + ((long)b * LSEQ + 1 + g) * DMODEL;
    float s = 0.f;
    #pragma unroll
    for (int e = 0; e < 8; ++e) {
        int d = lane + 32 * e;
        s += hr[d] * head_W[d];
    }
    for (int off = 16; off; off >>= 1) s += __shfl_down(s, off, 32);
    if (lane == 0) out[row] = s + head_b[0];
}

// ---------------------------------------------------------------
extern "C" void kernel_launch(void* const* d_in, const int* in_sizes, int n_in,
                              void* d_out, int out_size, void* d_ws, size_t ws_size,
                              hipStream_t stream)
{
    (void)in_sizes; (void)n_in; (void)out_size; (void)ws_size;
    const float* gene_emb     = (const float*)d_in[0];
    const float* chr_emb      = (const float*)d_in[1];
    const float* locus_W      = (const float*)d_in[2];
    const float* locus_b      = (const float*)d_in[3];
    const float* pert_emb     = (const float*)d_in[4];
    const float* cond_W       = (const float*)d_in[5];
    const float* cond_b       = (const float*)d_in[6];
    const float* inp_W        = (const float*)d_in[7];
    const float* inp_b        = (const float*)d_in[8];
    const float* head_W       = (const float*)d_in[9];
    const float* head_b       = (const float*)d_in[10];
    const int*   chr_idx      = (const int*)d_in[11];
    const float* locus_fourier= (const float*)d_in[12];
    const float* pathway      = (const float*)d_in[13];
    const int*   pert_idx     = (const int*)d_in[14];
    const float* mW_in[2]   = {(const float*)d_in[15], (const float*)d_in[23]};
    const float* mconv_w[2] = {(const float*)d_in[16], (const float*)d_in[24]};
    const float* mconv_b[2] = {(const float*)d_in[17], (const float*)d_in[25]};
    const float* mdt_bias[2]= {(const float*)d_in[18], (const float*)d_in[26]};
    const float* mA_log[2]  = {(const float*)d_in[19], (const float*)d_in[27]};
    const float* mD[2]      = {(const float*)d_in[20], (const float*)d_in[28]};
    const float* mnorm[2]   = {(const float*)d_in[21], (const float*)d_in[29]};
    const float* mW_out[2]  = {(const float*)d_in[22], (const float*)d_in[30]};

    size_t off = 0;
    auto alloc = [&](size_t bytes) -> char* {
        char* r = (char*)d_ws + off;
        off += (bytes + 255) & ~(size_t)255;
        return r;
    };
    bf16_t* featsB = (bf16_t*)alloc((size_t)NTOK * GFDP * 2);
    bf16_t* inpWT  = (bf16_t*)alloc((size_t)DMODEL * GFDP * 2);
    bf16_t* EB     = (bf16_t*)alloc((size_t)NTOK * DMODEL * 2);
    bf16_t* WinT   = (bf16_t*)alloc((size_t)DINP * DMODEL * 2);
    bf16_t* WoutT  = (bf16_t*)alloc((size_t)DMODEL * DI * 2);
    float*  Z      = (float*) alloc((size_t)NTOK * DINP * 4);
    float*  xbuf   = (float*) alloc((size_t)LHAT * DI * 4);
    bf16_t* Bmb    = (bf16_t*)alloc((size_t)LHAT * NSD * 2);
    bf16_t* Cmb    = (bf16_t*)alloc((size_t)LHAT * NSD * 2);
    bf16_t* BmT    = (bf16_t*)alloc((size_t)NSD * LHAT * 2);
    float*  dtb    = (float*) alloc((size_t)LHAT * HH * 4);
    float*  ab     = (float*) alloc((size_t)LHAT * HH * 4);
    float*  cumA   = (float*) alloc((size_t)LHAT * HH * 4);
    float*  csum   = (float*) alloc((size_t)NCHUNK * HH * 4);
    bf16_t* xdtT   = (bf16_t*)alloc((size_t)DI * LHAT * 2);
    bf16_t* xwT    = (bf16_t*)alloc((size_t)DI * LHAT * 2);
    bf16_t* Gt     = (bf16_t*)alloc((size_t)NCHUNK * HH * QCH * QCH * 2);
    float*  Sloc   = (float*) alloc((size_t)NCHUNK * HH * PP * NSD * 4);
    bf16_t* HprevB = (bf16_t*)alloc((size_t)NCHUNK * HH * PP * NSD * 2);
    float*  ybuf   = (float*) alloc((size_t)LHAT * DI * 4);
    bf16_t* gn     = (bf16_t*)alloc((size_t)LHAT * DI * 2);
    float*  Hc     = (float*) alloc((size_t)8 * LSEQ * DMODEL * 4);

    auto cdiv = [](long a, long b) { return (int)((a + b - 1) / b); };

    // 1) features (shared genes + 8 cond rows)
    k_build_feats<<<NTOK, 64, 0, stream>>>(gene_emb, chr_emb, locus_W, locus_b,
                                           pert_emb, cond_W, cond_b, chr_idx,
                                           locus_fourier, pathway, pert_idx, featsB);

    // 2) embed GEMM: E = feats @ inp_W + inp_b  (bf16 out only; K=192 -> no TDM pad)
    k_transpose_pad<<<cdiv((long)DMODEL * GFDP, 256), 256, 0, stream>>>(
        inp_W, inpWT, 178, DMODEL, GFDP, DMODEL);
    k_gemm<<<dim3(cdiv(NTOK / 32, 8), DMODEL / 32), 256, 0, stream>>>(
        featsB, GFDP, inpWT, GFDP, inp_b,
        (float*)nullptr, 0, EB, DMODEL,
        NTOK, DMODEL, GFDP, 0, NTOK, 0, 0);

    for (int dir = 0; dir < 2; ++dir) {
        // 3) in-projection GEMM (batch-independent): Z = E @ W_in   (TDM B-panel)
        k_transpose_pad<<<cdiv((long)DINP * DMODEL, 256), 256, 0, stream>>>(
            mW_in[dir], WinT, DMODEL, DINR, DMODEL, DINP);
        k_transpose_pad<<<cdiv((long)DMODEL * DI, 256), 256, 0, stream>>>(
            mW_out[dir], WoutT, DI, DMODEL, DI, DMODEL);
        k_gemm<<<dim3(cdiv(NTOK / 32, 8), DINP / 32), 256,
                 32 * (DMODEL + 8) * 2, stream>>>(
            EB, DMODEL, WinT, DMODEL, (const float*)nullptr,
            Z, DINP, (bf16_t*)nullptr, 0,
            NTOK, DINP, DMODEL, 0, NTOK, 0, 1);

        for (int b = 0; b < 8; ++b) {
            k_conv_split<<<LHAT, 256, 0, stream>>>(Z, mconv_w[dir], mconv_b[dir],
                                                   mdt_bias[dir], mA_log[dir],
                                                   xbuf, Bmb, Cmb, BmT, dtb, ab, dir, b);
            k_cum_a<<<NCHUNK * HH, QCH, 0, stream>>>(ab, cumA, csum);
            k_weight_x<<<LHAT, 256, 0, stream>>>(xbuf, dtb, cumA, csum, xdtT, xwT);
            // masked decay attention matrix (per chunk, per head)
            k_gmat<<<dim3(2, HH, NCHUNK), 256, 0, stream>>>(Cmb, Bmb, cumA, dtb, Gt);
            // chunk-local states: S[c,h] = xw^T @ B   (64 x 128, K=128)
            k_gemm_batched<<<dim3(1, HH, NCHUNK), 256, 0, stream>>>(
                xwT, LHAT, (long)PP * LHAT, (long)QCH,
                BmT, LHAT, 0L, (long)QCH,
                Sloc, NSD, (long)PP * NSD, (long)HH * PP * NSD,
                PP, NSD, QCH);
            k_recur<<<HH, 256, 0, stream>>>(Sloc, csum, HprevB);
            // intra-chunk: y = G~ @ (dt*x)   (128 x 64, K=128)
            k_gemm_batched<<<dim3(1, HH, NCHUNK), 256, 0, stream>>>(
                Gt, QCH, (long)QCH * QCH, (long)HH * QCH * QCH,
                xdtT, LHAT, (long)PP * LHAT, (long)QCH,
                ybuf, DI, (long)PP, (long)QCH * DI,
                QCH, PP, QCH);
            // inter-chunk contribution (+=)
            k_inter<<<dim3(1, HH, NCHUNK), 256, 0, stream>>>(Cmb, HprevB, cumA, ybuf);
            // gate + RMSNorm -> bf16 rows (zeros in pad region)
            k_gate<<<LHAT, 256, 0, stream>>>(ybuf, xbuf, Z, mD[dir], mnorm[dir],
                                             gn, dir, b);
            // out-projection into Hc[b] with fwd/flip row mapping; bwd accumulates
            k_gemm<<<dim3(cdiv(LHAT / 32, 8), DMODEL / 32), 256,
                     32 * (DI + 8) * 2, stream>>>(
                gn, DI, WoutT, DI, (const float*)nullptr,
                Hc + (long)b * LSEQ * DMODEL, DMODEL, (bf16_t*)nullptr, 0,
                LHAT, DMODEL, DI, dir ? 1 : 0, LSEQ, dir ? 1 : 0, 1);
        }
    }

    // 4) head
    k_head<<<8192, 256, 0, stream>>>(Hc, head_W, head_b, (float*)d_out);
}
